// MatrixGCN_9801115369777
// MI455X (gfx1250) — matrix operand order repaired and checked by reading, not compiled, not run on hardware
//
#include <hip/hip_runtime.h>
#include <hip/hip_bf16.h>

// ---------------------------------------------------------------------------
// MatrixGCN for MI455X (gfx1250, wave32, WMMA)
//   conv1d -> bf16 -> WMMA GEMM1 -> norm-aggregate+bias -> relu -> bf16
//          -> WMMA GEMM2 -> norm-aggregate+bias -> d_out (f32)
// ---------------------------------------------------------------------------

typedef __attribute__((ext_vector_type(16))) __bf16 v16bf;
typedef __attribute__((ext_vector_type(8)))  float  v8f;

#define NNODES 20000
#define NEDGES 320000
#define IN_CH  512
#define HID    256
#define EMB    128

// ---------------- Conv1d(1,64,24,stride=24) on transposed input ------------
// h0[n, o*8+w] = cb[o] + sum_t x[n, t, w] * cw[o, t]   (stored as bf16)
__global__ void conv_node_kernel(const float* __restrict__ x,
                                 const float* __restrict__ cw,
                                 const float* __restrict__ cb,
                                 __hip_bfloat16* __restrict__ h0,
                                 int n_nodes) {
  int idx = blockIdx.x * blockDim.x + threadIdx.x;   // n*512 + f
  if (idx >= n_nodes * 512) return;
  int n = idx >> 9;
  int f = idx & 511;
  int o = f >> 3;
  int w = f & 7;
  const float* xp = x + (size_t)n * 192 + w;  // x[n, t, w], stride 8 over t
  const float* wp = cw + o * 24;
  float acc = cb[o];
#pragma unroll
  for (int t = 0; t < 24; ++t) acc = fmaf(xp[t * 8], wp[t], acc);
  h0[idx] = __float2bfloat16(acc);
}

// ---------------- weight transpose + f32 -> bf16 ---------------------------
// W [K, Nd] row-major  ->  Wt [Nd, K] row-major (bf16)
__global__ void wt_bf16_kernel(const float* __restrict__ W,
                               __hip_bfloat16* __restrict__ Wt,
                               int K, int Nd) {
  int i = blockIdx.x * blockDim.x + threadIdx.x;
  if (i >= K * Nd) return;
  int k = i / Nd;
  int n = i - k * Nd;
  Wt[(size_t)n * K + k] = __float2bfloat16(W[i]);
}

// ---------------- degree / dinv --------------------------------------------
__global__ void deg_init_kernel(float* __restrict__ deg, int n) {
  int i = blockIdx.x * blockDim.x + threadIdx.x;
  if (i < n) deg[i] = 1.0f;                      // self loop
}
__global__ void deg_accum_kernel(const int* __restrict__ dst,
                                 float* __restrict__ deg, int e) {
  int i = blockIdx.x * blockDim.x + threadIdx.x;
  if (i < e) atomicAdd(&deg[dst[i]], 1.0f);
}
__global__ void deg_rsqrt_kernel(float* __restrict__ deg, int n) {
  int i = blockIdx.x * blockDim.x + threadIdx.x;
  if (i < n) deg[i] = rsqrtf(deg[i]);            // deg >= 1 always
}

// ---------------- WMMA GEMM: C[M,ND] = A[M,KD] (bf16) * Bt[ND,KD]^T --------
// One wave per 16x16 tile; 8 waves/block -> 16 rows x 128 cols per block.
template <int KD, int ND>
__global__ void wmma_gemm_kernel(const __hip_bfloat16* __restrict__ A,
                                 const __hip_bfloat16* __restrict__ Bt,
                                 float* __restrict__ C) {
  const int lane = threadIdx.x & 31;
  const int wave = threadIdx.x >> 5;
  const int m0 = blockIdx.x * 16;
  const int n0 = (blockIdx.y * 8 + wave) * 16;
  if (n0 >= ND) return;

  const int half = lane >> 4;        // 0: lanes 0-15, 1: lanes 16-31
  const int l    = lane & 15;

  union Frag { v16bf v; uint4 q[2]; };

  // A fragment: row m0+l; lanes 0-15 take K = kb+0..7 & kb+16..23,
  //             lanes 16-31 take K = kb+8..15 & kb+24..31 (ISA 7.12.2).
  const __hip_bfloat16* aptr = A  + (size_t)(m0 + l) * KD + half * 8;
  // B fragment via transposed weights: col n0+l; lanes 0-15 K=kb+0..7 & kb+16..23,
  //             lanes 16-31 K=kb+8..15 & kb+24..31 (same order as A).
  const __hip_bfloat16* bptr = Bt + (size_t)(n0 + l) * KD + half * 8;

  v8f acc = {};
  for (int kb = 0; kb < KD; kb += 32) {
    Frag a, b;
    a.q[0] = *(const uint4*)(aptr + kb);        // 8 bf16: K pairs 0..3
    a.q[1] = *(const uint4*)(aptr + kb + 16);   // 8 bf16: K pairs 8..11
    b.q[0] = *(const uint4*)(bptr + kb);
    b.q[1] = *(const uint4*)(bptr + kb + 16);
    if (kb + 32 < KD) {
      __builtin_prefetch(aptr + kb + 32, 0, 0);
      __builtin_prefetch(bptr + kb + 32, 0, 0);
    }
    acc = __builtin_amdgcn_wmma_f32_16x16x32_bf16(
        /*neg_a=*/false, a.v, /*neg_b=*/false, b.v,
        /*c_mod=*/(short)0, acc, /*reuse_a=*/false, /*reuse_b=*/false);
  }

  // C/D layout: VGPR j -> M = j + 8*half, N = l.
  float* crow = C + (size_t)(m0 + half * 8) * ND + n0 + l;
#pragma unroll
  for (int j = 0; j < 8; ++j) crow[(size_t)j * ND] = acc[j];
}

// ---------------- aggregation init: self loop + bias -----------------------
// out[n,c] = dinv[n]^2 * hw[n,c] + bias[c]
__global__ void agg_init_kernel(const float* __restrict__ hw,
                                const float* __restrict__ dinv,
                                const float* __restrict__ bias,
                                float* __restrict__ out,
                                int C, int n_nodes) {
  int i = blockIdx.x * blockDim.x + threadIdx.x;
  if (i >= n_nodes * C) return;
  int n = i / C;
  int c = i - n * C;
  float dv = dinv[n];
  out[i] = dv * dv * hw[i] + bias[c];
}

// ---------------- per-edge scatter: out[d] += dinv[s]*dinv[d]*hw[s] --------
// One block per edge, one thread per channel (blockDim.x == C).
__global__ void edge_scatter_kernel(const int* __restrict__ src,
                                    const int* __restrict__ dst,
                                    const float* __restrict__ dinv,
                                    const float* __restrict__ hw,
                                    float* __restrict__ out,
                                    int C, int n_edges) {
  int e = blockIdx.x;
  if (e >= n_edges) return;
  int s = src[e];
  int d = dst[e];
  float nrm = dinv[s] * dinv[d];
  int c = threadIdx.x;
  atomicAdd(&out[(size_t)d * C + c], nrm * hw[(size_t)s * C + c]);
}

// ---------------- relu + f32 -> bf16 ---------------------------------------
__global__ void relu_bf16_kernel(const float* __restrict__ in,
                                 __hip_bfloat16* __restrict__ out, int total) {
  int i = blockIdx.x * blockDim.x + threadIdx.x;
  if (i < total) out[i] = __float2bfloat16(fmaxf(in[i], 0.0f));
}

// ---------------------------------------------------------------------------
static inline size_t align256(size_t x) { return (x + 255) & ~(size_t)255; }

extern "C" void kernel_launch(void* const* d_in, const int* in_sizes, int n_in,
                              void* d_out, int out_size, void* d_ws, size_t ws_size,
                              hipStream_t stream) {
  const float* x    = (const float*)d_in[0];
  const int*   ei   = (const int*)d_in[1];     // [2, E] int32
  const float* cw   = (const float*)d_in[2];
  const float* cb   = (const float*)d_in[3];
  const float* W1   = (const float*)d_in[4];
  const float* b1   = (const float*)d_in[5];
  const float* W2   = (const float*)d_in[6];
  const float* b2   = (const float*)d_in[7];
  float*       out  = (float*)d_out;

  const int n = in_sizes[0] / 192;             // 20000
  const int e = in_sizes[1] / 2;               // 320000
  const int* src = ei;
  const int* dst = ei + e;

  // -------- workspace carve-up --------
  char* p = (char*)d_ws;
  __hip_bfloat16* h0bf = (__hip_bfloat16*)p; p += align256((size_t)n * IN_CH * 2);
  __hip_bfloat16* w1t  = (__hip_bfloat16*)p; p += align256((size_t)IN_CH * HID * 2);
  __hip_bfloat16* w2t  = (__hip_bfloat16*)p; p += align256((size_t)HID * EMB * 2);
  float*          dinv = (float*)p;          p += align256((size_t)n * 4);
  float*          hw1  = (float*)p;          p += align256((size_t)n * HID * 4);
  float*          o1   = (float*)p;          p += align256((size_t)n * HID * 4);
  __hip_bfloat16* h1bf = (__hip_bfloat16*)p; p += align256((size_t)n * HID * 2);
  float*          hw2  = (float*)p;          p += align256((size_t)n * EMB * 4);
  (void)ws_size;

  const int TPB = 256;

  // 1) conv1d -> h0 (bf16)  [n, 512]
  conv_node_kernel<<<(n * IN_CH + TPB - 1) / TPB, TPB, 0, stream>>>(
      x, cw, cb, h0bf, n);

  // 2) weight transposes (bf16)
  wt_bf16_kernel<<<(IN_CH * HID + TPB - 1) / TPB, TPB, 0, stream>>>(
      W1, w1t, IN_CH, HID);
  wt_bf16_kernel<<<(HID * EMB + TPB - 1) / TPB, TPB, 0, stream>>>(
      W2, w2t, HID, EMB);

  // 3) degree -> dinv (in place)
  deg_init_kernel<<<(n + TPB - 1) / TPB, TPB, 0, stream>>>(dinv, n);
  deg_accum_kernel<<<(e + TPB - 1) / TPB, TPB, 0, stream>>>(dst, dinv, e);
  deg_rsqrt_kernel<<<(n + TPB - 1) / TPB, TPB, 0, stream>>>(dinv, n);

  // 4) GEMM1: hw1[n,256] = h0 @ W1  (WMMA bf16)
  {
    dim3 grid(n / 16, HID / 128);
    wmma_gemm_kernel<IN_CH, HID><<<grid, 256, 0, stream>>>(h0bf, w1t, hw1);
  }

  // 5) aggregate layer 1: o1 = selfloop + bias, then edge atomics
  agg_init_kernel<<<(n * HID + TPB - 1) / TPB, TPB, 0, stream>>>(
      hw1, dinv, b1, o1, HID, n);
  edge_scatter_kernel<<<e, HID, 0, stream>>>(src, dst, dinv, hw1, o1, HID, e);

  // 6) relu -> bf16 activations for layer 2
  relu_bf16_kernel<<<(n * HID + TPB - 1) / TPB, TPB, 0, stream>>>(
      o1, h1bf, n * HID);

  // 7) GEMM2: hw2[n,128] = h1 @ W2  (WMMA bf16)
  {
    dim3 grid(n / 16, 1);
    wmma_gemm_kernel<HID, EMB><<<grid, 256, 0, stream>>>(h1bf, w2t, hw2);
  }

  // 8) aggregate layer 2 straight into d_out (full init, then atomics)
  agg_init_kernel<<<(n * EMB + TPB - 1) / TPB, TPB, 0, stream>>>(
      hw2, dinv, b2, out, EMB, n);
  edge_scatter_kernel<<<e, EMB, 0, stream>>>(src, dst, dinv, hw2, out, EMB, e);

  (void)n_in; (void)out_size;
}
